// CausalGCN_8993661518245
// MI455X (gfx1250) — compile-verified, matrix operand order checked by reading, not run on hardware
//
#include <hip/hip_runtime.h>
#include <hip/hip_bf16.h>

typedef __attribute__((ext_vector_type(16))) _Float16 v16h;
typedef __attribute__((ext_vector_type(8)))  _Float16 v8h;
typedef __attribute__((ext_vector_type(8)))  float    v8f;

static const int NN   = 50000;
static const int EE   = 400000;
static const int HH   = 128;
static const int GG   = 512;
static const int CO   = 10;
static const int LL   = 3;

// ---------------------------------------------------------------- helpers
static __device__ __forceinline__ float wave_red32(float v) {
#pragma unroll
  for (int o = 16; o > 0; o >>= 1) v += __shfl_xor(v, o, 32);
  return v;
}
static __device__ __forceinline__ float elu1(float v) {
  return v > 0.f ? v : (expf(v) - 1.f);
}

// ---------------------------------------------------------------- fill
__global__ void fill_kernel(float* p, float v, long n) {
  long i = (long)blockIdx.x * blockDim.x + threadIdx.x;
  if (i < n) p[i] = v;
}

// ---------------------------------------------------------------- BatchNorm
// partial column sums: blockDim must be 256, C must divide 256 (C in {128,256})
__global__ void bn_partial_kernel(const float* __restrict__ x, float* __restrict__ acc,
                                  int n, int C) {
  __shared__ float ssum[256];
  __shared__ float ssq[256];
  int t = threadIdx.x;
  int col = t % C;
  int copies = 256 / C;
  int cidx = t / C;
  float s = 0.f, q = 0.f;
  for (int r = blockIdx.x * copies + cidx; r < n; r += gridDim.x * copies) {
    float v = x[(size_t)r * C + col];
    s += v; q += v * v;
  }
  ssum[t] = s; ssq[t] = q;
  __syncthreads();
  if (t < C) {
    float S = ssum[t], Q = ssq[t];
    for (int c2 = 1; c2 < copies; ++c2) { S += ssum[t + c2 * C]; Q += ssq[t + c2 * C]; }
    atomicAdd(&acc[t], S);
    atomicAdd(&acc[C + t], Q);
  }
}

__global__ void bn_finalize_kernel(const float* __restrict__ acc, float* __restrict__ stats,
                                   int n, int C) {
  int c = blockIdx.x * blockDim.x + threadIdx.x;
  if (c >= C) return;
  float inv_n = 1.f / (float)n;
  float mean = acc[c] * inv_n;
  float var  = acc[C + c] * inv_n - mean * mean;
  if (var < 0.f) var = 0.f;
  stats[c]     = mean;
  stats[C + c] = rsqrtf(var + 1e-5f);
}

__global__ void bn_apply_kernel(const float* __restrict__ x, const float* __restrict__ stats,
                                float* __restrict__ o32, _Float16* __restrict__ o16,
                                long total, int C) {
  long i = (long)blockIdx.x * blockDim.x + threadIdx.x;
  if (i >= total) return;
  int c = (int)(i % C);
  float v = (x[i] - stats[c]) * stats[C + c] + 1e-4f;
  if (o32) o32[i] = v;
  if (o16) o16[i] = (_Float16)v;
}

// ---------------------------------------------------------------- weight fp32 -> fp16 transposed
// Wt[n*K + k] = (f16) W[k*NO + n]
__global__ void wconv_kernel(const float* __restrict__ W, _Float16* __restrict__ Wt,
                             int K, int NO) {
  int idx = blockIdx.x * blockDim.x + threadIdx.x;
  if (idx >= K * NO) return;
  int k = idx / NO, n = idx % NO;
  Wt[(size_t)n * K + k] = (_Float16)W[idx];
}

// ---------------------------------------------------------------- WMMA GEMM
// C[M,NO] = act( A16[M,K] * Bt16[NO,K]^T + bias ).  One wave per 16x16 tile.
// act: 0=none 1=relu 2=elu 3=elu(elu)
__global__ void wmma_gemm_kernel(const _Float16* __restrict__ A,
                                 const _Float16* __restrict__ Bt,
                                 const float* __restrict__ bias,
                                 float* __restrict__ C,
                                 int M, int K, int NO, int act) {
  int lane = threadIdx.x & 31;
  int m0 = blockIdx.x * 16;
  int n0 = blockIdx.y * 16;
  if (m0 >= M) return;
  int half = lane >> 4;          // 0: K low slice, 1: K high slice
  int l16  = lane & 15;
  int mA = m0 + l16; if (mA >= M) mA = M - 1;
  int nB = n0 + l16;
  int klo = half * 8;
  v8f acc = {};
  for (int k0 = 0; k0 < K; k0 += 32) {
    const _Float16* ap = A  + (size_t)mA * K + k0 + klo;
    const _Float16* bp = Bt + (size_t)nB * K + k0 + klo;
    v8h alo = *(const v8h*)(ap);
    v8h ahi = *(const v8h*)(ap + 16);
    v8h blo = *(const v8h*)(bp);
    v8h bhi = *(const v8h*)(bp + 16);
    v16h av, bv;
#pragma unroll
    for (int i = 0; i < 8; ++i) {
      av[i] = alo[i]; av[i + 8] = ahi[i];
      bv[i] = blo[i]; bv[i + 8] = bhi[i];
    }
    acc = __builtin_amdgcn_wmma_f32_16x16x32_f16(false, av, false, bv,
                                                 (short)0, acc, false, false);
  }
  int n = n0 + l16;
#pragma unroll
  for (int r = 0; r < 8; ++r) {
    int m = m0 + r + half * 8;
    if (m < M && n < NO) {
      float v = acc[r];
      if (bias) v += bias[n];
      if (act == 1)      v = v > 0.f ? v : 0.f;
      else if (act == 2) v = elu1(v);
      else if (act == 3) v = elu1(elu1(v));
      C[(size_t)m * NO + n] = v;
    }
  }
}

// ---------------------------------------------------------------- degree / norm
__global__ void deg_edge_kernel(const int* __restrict__ src, const float* __restrict__ ew,
                                int ews, float* __restrict__ deg, int E) {
  int e = blockIdx.x * blockDim.x + threadIdx.x;
  if (e >= E) return;
  float w = ew ? ew[(size_t)e * ews] : 1.f;
  atomicAdd(&deg[src[e]], w);
}
__global__ void dis_kernel(float* __restrict__ deg, int n) {
  int i = blockIdx.x * blockDim.x + threadIdx.x;
  if (i >= n) return;
  float d = deg[i];
  deg[i] = d > 0.f ? rsqrtf(d) : 0.f;
}
__global__ void edge_coef_kernel(const int* __restrict__ src, const int* __restrict__ dst,
                                 const float* __restrict__ ew, int ews,
                                 const float* __restrict__ dis, float* __restrict__ coef, int E) {
  int e = blockIdx.x * blockDim.x + threadIdx.x;
  if (e >= E) return;
  float w = ew ? ew[(size_t)e * ews] : 1.f;
  coef[e] = dis[src[e]] * w * dis[dst[e]];
}

// ---------------------------------------------------------------- conv aggregation
__global__ void conv_init_kernel(const float* __restrict__ hW, const float* __restrict__ b,
                                 const float* __restrict__ dis, float* __restrict__ out,
                                 long n, int H) {
  long idx = (long)blockIdx.x * blockDim.x + threadIdx.x;
  if (idx >= n * H) return;
  long i = idx / H; int f = (int)(idx % H);
  float d = dis[i];
  out[idx] = b[f] + d * d * hW[idx];
}
__global__ void conv_edge_kernel(const float* __restrict__ hW, const int* __restrict__ src,
                                 const int* __restrict__ dst, const float* __restrict__ coef,
                                 float* __restrict__ out, long E, int H) {
  long idx = (long)blockIdx.x * blockDim.x + threadIdx.x;
  if (idx >= E * H) return;
  long e = idx / H; int f = (int)(idx % H);
  atomicAdd(&out[(size_t)dst[e] * H + f], coef[e] * hW[(size_t)src[e] * H + f]);
}

__global__ void act_kernel(float* __restrict__ p, long n, int mode) {
  long i = (long)blockIdx.x * blockDim.x + threadIdx.x;
  if (i >= n) return;
  float v = p[i];
  p[i] = (mode == 1) ? (v > 0.f ? v : 0.f) : elu1(v);
}

// ---------------------------------------------------------------- edge attention (wave per edge)
__global__ void edge_att_kernel(const float* __restrict__ h, const float* __restrict__ eW,
                                const float* __restrict__ eb, const int* __restrict__ src,
                                const int* __restrict__ dst, float* __restrict__ eatt, int E) {
  int gw = (blockIdx.x * blockDim.x + threadIdx.x) >> 5;
  int lane = threadIdx.x & 31;
  if (gw >= E) return;
  int s = src[gw], d = dst[gw];
  float4 a = ((const float4*)(h + (size_t)s * HH))[lane];
  float4 b = ((const float4*)(h + (size_t)d * HH))[lane];
  int f = lane * 4;
  float p0 = a.x * eW[(f + 0) * 2] + a.y * eW[(f + 1) * 2] + a.z * eW[(f + 2) * 2] + a.w * eW[(f + 3) * 2];
  float p1 = a.x * eW[(f + 0) * 2 + 1] + a.y * eW[(f + 1) * 2 + 1] + a.z * eW[(f + 2) * 2 + 1] + a.w * eW[(f + 3) * 2 + 1];
  int g = 128 + f;
  p0 += b.x * eW[(g + 0) * 2] + b.y * eW[(g + 1) * 2] + b.z * eW[(g + 2) * 2] + b.w * eW[(g + 3) * 2];
  p1 += b.x * eW[(g + 0) * 2 + 1] + b.y * eW[(g + 1) * 2 + 1] + b.z * eW[(g + 2) * 2 + 1] + b.w * eW[(g + 3) * 2 + 1];
  p0 = wave_red32(p0);
  p1 = wave_red32(p1);
  if (lane == 0) {
    float z0 = p0 + eb[0], z1 = p1 + eb[1];
    float m = fmaxf(z0, z1);
    float e0 = expf(z0 - m), e1 = expf(z1 - m);
    float inv = 1.f / (e0 + e1);
    eatt[(size_t)gw * 2 + 0] = e0 * inv;
    eatt[(size_t)gw * 2 + 1] = e1 * inv;
  }
}

// ---------------------------------------------------------------- node 2-wide linear (wave per node)
__global__ void node_lin2_kernel(const float* __restrict__ h, const float* __restrict__ W2,
                                 float* __restrict__ out2, int n) {
  int gw = (blockIdx.x * blockDim.x + threadIdx.x) >> 5;
  int lane = threadIdx.x & 31;
  if (gw >= n) return;
  float4 a = ((const float4*)(h + (size_t)gw * HH))[lane];
  int f = lane * 4;
  float p0 = a.x * W2[(f + 0) * 2] + a.y * W2[(f + 1) * 2] + a.z * W2[(f + 2) * 2] + a.w * W2[(f + 3) * 2];
  float p1 = a.x * W2[(f + 0) * 2 + 1] + a.y * W2[(f + 1) * 2 + 1] + a.z * W2[(f + 2) * 2 + 1] + a.w * W2[(f + 3) * 2 + 1];
  p0 = wave_red32(p0);
  p1 = wave_red32(p1);
  if (lane == 0) {
    out2[(size_t)gw * 2 + 0] = p0;
    out2[(size_t)gw * 2 + 1] = p1;
  }
}

__global__ void softmax2_kernel(float* __restrict__ p, int n) {
  int i = blockIdx.x * blockDim.x + threadIdx.x;
  if (i >= n) return;
  float z0 = p[(size_t)i * 2], z1 = p[(size_t)i * 2 + 1];
  float m = fmaxf(z0, z1);
  float e0 = expf(z0 - m), e1 = expf(z1 - m);
  float inv = 1.f / (e0 + e1);
  p[(size_t)i * 2] = e0 * inv;
  p[(size_t)i * 2 + 1] = e1 * inv;
}

// out[i,f] = natt[i,ch] * h[i,f]
__global__ void scale_mul_kernel(const float* __restrict__ h, const float* __restrict__ natt,
                                 int ch, float* __restrict__ out, long n) {
  long idx = (long)blockIdx.x * blockDim.x + threadIdx.x;
  if (idx >= n * HH) return;
  long i = idx / HH;
  out[idx] = natt[(size_t)i * 2 + ch] * h[idx];
}

// pool[batch[i], f] += elu(x[i,f])
__global__ void elu_pool_kernel(const float* __restrict__ x, const int* __restrict__ batch,
                                float* __restrict__ pool, long n) {
  long idx = (long)blockIdx.x * blockDim.x + threadIdx.x;
  if (idx >= n * HH) return;
  long i = idx / HH; int f = (int)(idx % HH);
  atomicAdd(&pool[(size_t)batch[i] * HH + f], elu1(x[idx]));
}

__global__ void concat_kernel(const float* __restrict__ a, const float* __restrict__ b,
                              float* __restrict__ o, int rows) {
  int idx = blockIdx.x * blockDim.x + threadIdx.x;
  if (idx >= rows * 2 * HH) return;
  int r = idx / (2 * HH), c = idx % (2 * HH);
  o[idx] = (c < HH) ? a[(size_t)r * HH + c] : b[(size_t)r * HH + (c - HH)];
}

// small GEMM: C[M,NO] = A[M,K] * W[K,NO] + b
__global__ void gemm_small_kernel(const float* __restrict__ A, const float* __restrict__ W,
                                  const float* __restrict__ b, float* __restrict__ C,
                                  int M, int K, int NO) {
  int idx = blockIdx.x * blockDim.x + threadIdx.x;
  if (idx >= M * NO) return;
  int r = idx / NO, c = idx % NO;
  float s = b ? b[c] : 0.f;
  for (int k = 0; k < K; ++k) s += A[(size_t)r * K + k] * W[(size_t)k * NO + c];
  C[idx] = s;
}

__global__ void log_softmax_kernel(const float* __restrict__ in, float* __restrict__ out,
                                   int M, int NO) {
  int r = blockIdx.x * blockDim.x + threadIdx.x;
  if (r >= M) return;
  const float* p = in + (size_t)r * NO;
  float mx = p[0];
  for (int c = 1; c < NO; ++c) mx = fmaxf(mx, p[c]);
  float s = 0.f;
  for (int c = 0; c < NO; ++c) s += expf(p[c] - mx);
  float lse = mx + logf(s);
  for (int c = 0; c < NO; ++c) out[(size_t)r * NO + c] = p[c] - lse;
}

// ---------------------------------------------------------------- host side
static inline int gridFor(long n, int b) { return (int)((n + b - 1) / b); }

extern "C" void kernel_launch(void* const* d_in, const int* in_sizes, int n_in,
                              void* d_out, int out_size, void* d_ws, size_t ws_size,
                              hipStream_t stream) {
  (void)in_sizes; (void)n_in; (void)out_size; (void)ws_size;
  const float* x      = (const float*)d_in[0];
  const float* W_feat = (const float*)d_in[1];
  const float* conv_Ws= (const float*)d_in[2];
  const float* conv_bs= (const float*)d_in[3];
  const float* eW     = (const float*)d_in[4];
  const float* eb     = (const float*)d_in[5];
  const float* naW    = (const float*)d_in[6];
  const float* nab    = (const float*)d_in[7];
  const float* xcW    = (const float*)d_in[8];
  const float* xcb    = (const float*)d_in[9];
  const float* xoW    = (const float*)d_in[10];
  const float* xob    = (const float*)d_in[11];
  const float* cW1    = (const float*)d_in[12];
  const float* cb1    = (const float*)d_in[13];
  const float* cW2    = (const float*)d_in[14];
  const float* cb2    = (const float*)d_in[15];
  const float* oW1    = (const float*)d_in[16];
  const float* ob1    = (const float*)d_in[17];
  const float* oW2    = (const float*)d_in[18];
  const float* ob2    = (const float*)d_in[19];
  const float* coW1   = (const float*)d_in[20];
  const float* cob1   = (const float*)d_in[21];
  const float* coW2   = (const float*)d_in[22];
  const float* cob2   = (const float*)d_in[23];
  const int*   esrc   = (const int*)d_in[24];
  const int*   edst   = (const int*)d_in[25];
  const int*   batch  = (const int*)d_in[26];
  float* out = (float*)d_out;

  // ---- workspace carve-up
  char* wp = (char*)d_ws;
  auto carve = [&](size_t bytes) -> void* {
    void* p = wp; wp += ((bytes + 255) & ~(size_t)255); return p;
  };
  float*    h     = (float*)carve((size_t)NN * HH * 4);
  float*    t0    = (float*)carve((size_t)NN * HH * 4);
  float*    t1    = (float*)carve((size_t)NN * HH * 4);
  _Float16* a16   = (_Float16*)carve((size_t)NN * HH * 2);
  float*    dis1  = (float*)carve((size_t)NN * 4);
  float*    disX  = (float*)carve((size_t)NN * 4);
  float*    coef1 = (float*)carve((size_t)EE * 4);
  float*    coefX = (float*)carve((size_t)EE * 4);
  float*    eatt  = (float*)carve((size_t)EE * 2 * 4);
  float*    hn2   = (float*)carve((size_t)NN * 2 * 4);
  float*    natt  = (float*)carve((size_t)NN * 2 * 4);
  float*    acc   = (float*)carve(512 * 4);
  float*    stats = (float*)carve(512 * 4);
  float*    poolc = (float*)carve((size_t)GG * HH * 4);
  float*    poolo = (float*)carve((size_t)GG * HH * 4);
  float*    cocat = (float*)carve((size_t)GG * 2 * HH * 4);
  float*    hg    = (float*)carve((size_t)GG * HH * 4);
  float*    hf    = (float*)carve((size_t)GG * 2 * HH * 4);
  float*    lg    = (float*)carve((size_t)GG * CO * 4);
  _Float16* wt    = (_Float16*)carve((size_t)(8 * HH * HH + 2 * HH * HH) * 2);
  _Float16* wtf    = wt;
  _Float16* wtconv = wt + 1 * HH * HH;   // 3 matrices
  _Float16* wtxc   = wt + 4 * HH * HH;
  _Float16* wtxo   = wt + 5 * HH * HH;
  _Float16* wtc1   = wt + 6 * HH * HH;
  _Float16* wto1   = wt + 7 * HH * HH;
  _Float16* wtco1  = wt + 8 * HH * HH;   // 256x128

  const int B = 256;

  auto run_bn = [&](const float* xin, int n, int C, float* o32, _Float16* o16) {
    fill_kernel<<<gridFor(2 * C, B), B, 0, stream>>>(acc, 0.f, 2 * C);
    int copies = 256 / C;
    int g = (n + copies - 1) / copies; if (g > 240) g = 240; if (g < 1) g = 1;
    bn_partial_kernel<<<g, 256, 0, stream>>>(xin, acc, n, C);
    bn_finalize_kernel<<<1, 256, 0, stream>>>(acc, stats, n, C);
    bn_apply_kernel<<<gridFor((long)n * C, B), B, 0, stream>>>(xin, stats, o32, o16, (long)n * C, C);
  };
  auto run_dis = [&](const float* ew, int ews, float* dis) {
    fill_kernel<<<gridFor(NN, B), B, 0, stream>>>(dis, 1.f, NN); // self-loop weight
    deg_edge_kernel<<<gridFor(EE, B), B, 0, stream>>>(esrc, ew, ews, dis, EE);
    dis_kernel<<<gridFor(NN, B), B, 0, stream>>>(dis, NN);
  };
  auto run_gemm = [&](const _Float16* A, const _Float16* Bt, const float* bias,
                      float* Cout, int M, int K, int NO, int act) {
    dim3 g((M + 15) / 16, NO / 16);
    wmma_gemm_kernel<<<g, 32, 0, stream>>>(A, Bt, bias, Cout, M, K, NO, act);
  };

  // ---- weight conversion (fp32 -> fp16 transposed)
  wconv_kernel<<<gridFor(HH * HH, B), B, 0, stream>>>(W_feat, wtf, HH, HH);
  for (int i = 0; i < LL; ++i)
    wconv_kernel<<<gridFor(HH * HH, B), B, 0, stream>>>(conv_Ws + (size_t)i * HH * HH,
                                                        wtconv + (size_t)i * HH * HH, HH, HH);
  wconv_kernel<<<gridFor(HH * HH, B), B, 0, stream>>>(xcW, wtxc, HH, HH);
  wconv_kernel<<<gridFor(HH * HH, B), B, 0, stream>>>(xoW, wtxo, HH, HH);
  wconv_kernel<<<gridFor(HH * HH, B), B, 0, stream>>>(cW1, wtc1, HH, HH);
  wconv_kernel<<<gridFor(HH * HH, B), B, 0, stream>>>(oW1, wto1, HH, HH);
  wconv_kernel<<<gridFor(2 * HH * HH, B), B, 0, stream>>>(coW1, wtco1, 2 * HH, HH);

  // ---- symmetric norm for unit edge weights (shared by 3 convs + node-att conv)
  run_dis(nullptr, 0, dis1);
  edge_coef_kernel<<<gridFor(EE, B), B, 0, stream>>>(esrc, edst, nullptr, 0, dis1, coef1, EE);

  // ---- BN(x) -> relu(x @ W_feat)
  run_bn(x, NN, HH, nullptr, a16);
  run_gemm(a16, wtf, nullptr, h, NN, HH, HH, 1);

  // ---- L x [BN -> GCNConv -> relu]
  for (int i = 0; i < LL; ++i) {
    run_bn(h, NN, HH, nullptr, a16);
    run_gemm(a16, wtconv + (size_t)i * HH * HH, nullptr, t0, NN, HH, HH, 0);
    conv_init_kernel<<<gridFor((long)NN * HH, B), B, 0, stream>>>(t0, conv_bs + (size_t)i * HH, dis1, h, NN, HH);
    conv_edge_kernel<<<gridFor((long)EE * HH, B), B, 0, stream>>>(t0, esrc, edst, coef1, h, EE, HH);
    act_kernel<<<gridFor((long)NN * HH, B), B, 0, stream>>>(h, (long)NN * HH, 1);
  }

  // ---- edge attention
  edge_att_kernel<<<gridFor((long)EE * 32, B), B, 0, stream>>>(h, eW, eb, esrc, edst, eatt, EE);

  // ---- node attention: gcn_conv(h, naW, nab) then softmax
  node_lin2_kernel<<<gridFor((long)NN * 32, B), B, 0, stream>>>(h, naW, hn2, NN);
  conv_init_kernel<<<gridFor((long)NN * 2, B), B, 0, stream>>>(hn2, nab, dis1, natt, NN, 2);
  conv_edge_kernel<<<gridFor((long)EE * 2, B), B, 0, stream>>>(hn2, esrc, edst, coef1, natt, EE, 2);
  softmax2_kernel<<<gridFor(NN, B), B, 0, stream>>>(natt, NN);

  // ---- XC branch (channel 0)
  scale_mul_kernel<<<gridFor((long)NN * HH, B), B, 0, stream>>>(h, natt, 0, t0, NN);
  run_bn(t0, NN, HH, nullptr, a16);
  run_gemm(a16, wtxc, nullptr, t0, NN, HH, HH, 0);
  run_dis(eatt + 0, 2, disX);
  edge_coef_kernel<<<gridFor(EE, B), B, 0, stream>>>(esrc, edst, eatt + 0, 2, disX, coefX, EE);
  conv_init_kernel<<<gridFor((long)NN * HH, B), B, 0, stream>>>(t0, xcb, disX, t1, NN, HH);
  conv_edge_kernel<<<gridFor((long)EE * HH, B), B, 0, stream>>>(t0, esrc, edst, coefX, t1, EE, HH);
  fill_kernel<<<gridFor((long)GG * HH, B), B, 0, stream>>>(poolc, 0.f, (long)GG * HH);
  elu_pool_kernel<<<gridFor((long)NN * HH, B), B, 0, stream>>>(t1, batch, poolc, NN);

  // ---- XO branch (channel 1)
  scale_mul_kernel<<<gridFor((long)NN * HH, B), B, 0, stream>>>(h, natt, 1, t0, NN);
  run_bn(t0, NN, HH, nullptr, a16);
  run_gemm(a16, wtxo, nullptr, t0, NN, HH, HH, 0);
  run_dis(eatt + 1, 2, disX);
  edge_coef_kernel<<<gridFor(EE, B), B, 0, stream>>>(esrc, edst, eatt + 1, 2, disX, coefX, EE);
  conv_init_kernel<<<gridFor((long)NN * HH, B), B, 0, stream>>>(t0, xob, disX, t1, NN, HH);
  conv_edge_kernel<<<gridFor((long)EE * HH, B), B, 0, stream>>>(t0, esrc, edst, coefX, t1, EE, HH);
  fill_kernel<<<gridFor((long)GG * HH, B), B, 0, stream>>>(poolo, 0.f, (long)GG * HH);
  elu_pool_kernel<<<gridFor((long)NN * HH, B), B, 0, stream>>>(t1, batch, poolo, NN);

  // ---- C head
  run_bn(poolc, GG, HH, nullptr, a16);
  run_gemm(a16, wtc1, cb1, hg, GG, HH, HH, 1);
  run_bn(hg, GG, HH, hf, nullptr);
  gemm_small_kernel<<<gridFor(GG * CO, B), B, 0, stream>>>(hf, cW2, cb2, lg, GG, HH, CO);
  log_softmax_kernel<<<gridFor(GG, B), B, 0, stream>>>(lg, out, GG, CO);

  // ---- O head
  run_bn(poolo, GG, HH, nullptr, a16);
  run_gemm(a16, wto1, ob1, hg, GG, HH, HH, 1);
  run_bn(hg, GG, HH, hf, nullptr);
  gemm_small_kernel<<<gridFor(GG * CO, B), B, 0, stream>>>(hf, oW2, ob2, lg, GG, HH, CO);
  log_softmax_kernel<<<gridFor(GG, B), B, 0, stream>>>(lg, out + (size_t)GG * CO, GG, CO);

  // ---- CO head
  concat_kernel<<<gridFor((long)GG * 2 * HH, B), B, 0, stream>>>(poolc, poolo, cocat, GG);
  run_bn(cocat, GG, 2 * HH, nullptr, a16);
  run_gemm(a16, wtco1, cob1, hg, GG, 2 * HH, HH, 3);   // elu(elu(.))
  run_bn(hg, GG, HH, hf, nullptr);
  gemm_small_kernel<<<gridFor(GG * CO, B), B, 0, stream>>>(hf, coW2, cob2, lg, GG, HH, CO);
  log_softmax_kernel<<<gridFor(GG, B), B, 0, stream>>>(lg, out + (size_t)2 * GG * CO, GG, CO);
}
